// LlamaForSequenceRegression_14336600834254
// MI455X (gfx1250) — matrix operand order repaired and checked by reading, not compiled, not run on hardware
//
#include <hip/hip_runtime.h>
#include <hip/hip_bf16.h>
#include <math.h>
#include <stdint.h>

typedef __attribute__((ext_vector_type(16))) _Float16 v16h;
typedef __attribute__((ext_vector_type(4)))  _Float16 v4h;
typedef __attribute__((ext_vector_type(8)))  float    v8f;

#define BD   2
#define SD   1024
#define DD   2048
#define HH   16
#define HDD  128
#define FF   5632
#define RR   16
#define OUTD 11
#define MTOK (BD * SD)        // 2048 tokens
#define EPSF 1e-5f

// ---------------------------------------------------------------------------
// Embedding gather: h[tok, d] = embed[ids[tok], d]
// ---------------------------------------------------------------------------
__global__ void k_embed(const int* __restrict__ ids, const float* __restrict__ embed,
                        float* __restrict__ h) {
  long i = (long)blockIdx.x * blockDim.x + threadIdx.x;   // MTOK*DD threads
  long tok = i / DD, d = i % DD;
  h[i] = embed[(long)ids[tok] * DD + d];
}

// ---------------------------------------------------------------------------
// RMSNorm per token, fp32 in -> f16 out (WMMA operand precision)
// ---------------------------------------------------------------------------
__global__ __launch_bounds__(256)
void k_rmsnorm(const float* __restrict__ h, const float* __restrict__ g,
               _Float16* __restrict__ out) {
  int row = blockIdx.x;
  const float* x = h + (long)row * DD;
  __shared__ float red[256];
  float s = 0.f;
  for (int i = threadIdx.x; i < DD; i += 256) { float v = x[i]; s += v * v; }
  red[threadIdx.x] = s; __syncthreads();
  for (int o = 128; o > 0; o >>= 1) {
    if (threadIdx.x < o) red[threadIdx.x] += red[threadIdx.x + o];
    __syncthreads();
  }
  float rs = rsqrtf(red[0] * (1.0f / DD) + EPSF);
  for (int i = threadIdx.x; i < DD; i += 256)
    out[(long)row * DD + i] = (_Float16)(x[i] * rs * g[i]);
}

// ---------------------------------------------------------------------------
// WMMA GEMM: C[M,N] (+)= alpha * A[M,K](f16) @ B[K,N](f32->f16)
// 128x128 workgroup tile, 8 waves (2x4), each wave 64x32 = 4x2 WMMA frags.
// ALIGNED=true  : no guards; LDS double-buffered, async-DMA A tile, one
//                 barrier per K-step (stage t+1 while WMMAs consume t).
// ALIGNED=false : fully guarded single-buffer (tiny LoRA GEMMs, N=16/K=16).
// ---------------------------------------------------------------------------
#define TM 128
#define TN 128
#define TK 32
#define LDA_S 40     // halves; 80B row stride (16B aligned)
#define LDB_S 136    // halves; 272B row stride (8B aligned per 4-col group)

__device__ inline v16h load_a_frag(const _Float16* __restrict__ As, int mbase, int lane) {
  // CDNA5 16-bit A 16x32 layout: lanes 0-15 M=0..15 (K 0..15 over VGPR0-3,
  // K 16..31 over VGPR4-7); lanes 16-31 same M with K offset +8.
  int m  = mbase + (lane & 15);
  int hs = lane >> 4;
  v16h a;
#pragma unroll
  for (int e = 0; e < 16; ++e) {
    int j = e >> 1;
    int kk = ((j < 4) ? 0 : 16) + 2 * (j & 3) + 8 * hs + (e & 1);
    a[e] = As[m * LDA_S + kk];
  }
  return a;
}

__device__ inline v16h load_b_frag(const _Float16* __restrict__ Bs, int nbase, int lane) {
  // CDNA5 16-bit B 32x16 layout: lane&15 is the N column; halves 0-7 hold
  // K=8*hs+0..7, halves 8-15 hold K=16+8*hs+0..7 (same K order as A).
  int ncol = nbase + (lane & 15);
  int hsb = lane >> 4;
  v16h b;
#pragma unroll
  for (int e = 0; e < 16; ++e) b[e] = Bs[(e + 8 * hsb + ((e >= 8) ? 8 : 0)) * LDB_S + ncol];
  return b;
}

// Stage one K-tile: async-DMA the f16 A slab, load 4x float4 of fp32 weights,
// convert packed to f16, store 8B groups to LDS.
__device__ inline void stage_ab(const _Float16* asrc, unsigned ldsA,
                                const float* const* bsrc, _Float16* const* bdst) {
  asm volatile(
      "global_load_async_to_lds_b128 %0, %1, off\n\t"
      "global_load_async_to_lds_b128 %0, %1, off offset:16"
      :: "v"(ldsA), "v"(asrc) : "memory");
  float4 f0 = *(const float4*)bsrc[0];
  float4 f1 = *(const float4*)bsrc[1];
  float4 f2 = *(const float4*)bsrc[2];
  float4 f3 = *(const float4*)bsrc[3];
  v4h h0, h1, h2, h3;
  h0[0] = (_Float16)f0.x; h0[1] = (_Float16)f0.y; h0[2] = (_Float16)f0.z; h0[3] = (_Float16)f0.w;
  h1[0] = (_Float16)f1.x; h1[1] = (_Float16)f1.y; h1[2] = (_Float16)f1.z; h1[3] = (_Float16)f1.w;
  h2[0] = (_Float16)f2.x; h2[1] = (_Float16)f2.y; h2[2] = (_Float16)f2.z; h2[3] = (_Float16)f2.w;
  h3[0] = (_Float16)f3.x; h3[1] = (_Float16)f3.y; h3[2] = (_Float16)f3.z; h3[3] = (_Float16)f3.w;
  *(v4h*)bdst[0] = h0;
  *(v4h*)bdst[1] = h1;
  *(v4h*)bdst[2] = h2;
  *(v4h*)bdst[3] = h3;
}

template <bool ALIGNED>
__global__ __launch_bounds__(256)
void k_gemm_f16(const _Float16* __restrict__ A, int lda,
                const float* __restrict__ Bw, int ldb,
                float* __restrict__ C, int ldc,
                int M, int N, int K, float alpha, int accFlag) {
  __shared__ __align__(16) _Float16 As[2][TM * LDA_S];
  __shared__ __align__(16) _Float16 Bs[2][TK * LDB_S];

  const int tid  = threadIdx.x;
  const int lane = tid & 31;          // wave32
  const int wid  = tid >> 5;          // 8 waves
  const int wm   = wid >> 2;          // 0..1 -> 64-row stripes
  const int wn   = wid & 3;           // 0..3 -> 32-col stripes
  const int m0   = blockIdx.y * TM;
  const int n0   = blockIdx.x * TN;

  const int rowA  = tid >> 1;
  const int colhA = (tid & 1) * 16;

  v8f acc[4][2];
#pragma unroll
  for (int mi = 0; mi < 4; ++mi)
#pragma unroll
    for (int ni = 0; ni < 2; ++ni) { v8f z = {}; acc[mi][ni] = z; }

  if (ALIGNED) {
    // ---- hoisted per-thread staging pointers ----
    const _Float16* asrc = A + (long)(m0 + rowA) * lda + colhA;
    unsigned ldsA[2] = {
      (unsigned)(uintptr_t)&As[0][rowA * LDA_S + colhA],
      (unsigned)(uintptr_t)&As[1][rowA * LDA_S + colhA] };
    const float* bsrc[4];
    _Float16* bdst0[4];
    _Float16* bdst1[4];
#pragma unroll
    for (int c = 0; c < 4; ++c) {
      int idx = tid + c * 256;          // 0..1023
      int row = idx >> 5;               // 0..31
      int col = (idx & 31) * 4;         // 0..124
      bsrc[c]  = Bw + (long)row * ldb + n0 + col;
      bdst0[c] = &Bs[0][row * LDB_S + col];
      bdst1[c] = &Bs[1][row * LDB_S + col];
    }

    const int nT = K / TK;
    // prologue: stage tile 0 into buffer 0
    stage_ab(asrc, ldsA[0], bsrc, bdst0);
    asrc += TK;
#pragma unroll
    for (int c = 0; c < 4; ++c) bsrc[c] += (long)TK * ldb;
    asm volatile("s_wait_asynccnt 0" ::: "memory");
    __syncthreads();

    for (int t = 0; t < nT; ++t) {
      const int cur = t & 1;
      if (t + 1 < nT) {
        stage_ab(asrc, ldsA[cur ^ 1], bsrc, cur ? bdst0 : bdst1);
        asrc += TK;
#pragma unroll
        for (int c = 0; c < 4; ++c) bsrc[c] += (long)TK * ldb;
        if (t + 2 < nT)
          __builtin_prefetch(bsrc[0], 0, 1);
      }
      const _Float16* Asb = As[cur];
      const _Float16* Bsb = Bs[cur];
      v16h af[4], bf[2];
#pragma unroll
      for (int mi = 0; mi < 4; ++mi) af[mi] = load_a_frag(Asb, wm * 64 + mi * 16, lane);
#pragma unroll
      for (int ni = 0; ni < 2; ++ni) bf[ni] = load_b_frag(Bsb, wn * 32 + ni * 16, lane);
#pragma unroll
      for (int mi = 0; mi < 4; ++mi)
#pragma unroll
        for (int ni = 0; ni < 2; ++ni)
          acc[mi][ni] = __builtin_amdgcn_wmma_f32_16x16x32_f16(
              false, af[mi], false, bf[ni], (short)0, acc[mi][ni], false, false);
      if (t + 1 < nT) {
        asm volatile("s_wait_asynccnt 0" ::: "memory");
        __syncthreads();
      }
    }
  } else {
    // ---- guarded single-buffer path (LoRA shapes only) ----
    for (int kk = 0; kk < K; kk += TK) {
      {
        int gm = m0 + rowA;
        const _Float16* src = A + (long)gm * lda + kk + colhA;
        _Float16* dst = &As[0][rowA * LDA_S + colhA];
        if (gm < M && (kk + colhA + 16) <= K) {
          const uint4* s4 = (const uint4*)src;
          ((uint4*)dst)[0] = s4[0];
          ((uint4*)dst)[1] = s4[1];
        } else {
#pragma unroll
          for (int e = 0; e < 16; ++e)
            dst[e] = (gm < M && (kk + colhA + e) < K) ? src[e] : (_Float16)0.f;
        }
      }
#pragma unroll
      for (int c = 0; c < 4; ++c) {
        int idx = tid + c * 256;
        int row = idx >> 5;
        int col = (idx & 31) * 4;
        int gk = kk + row, gn = n0 + col;
        _Float16* dst = &Bs[0][row * LDB_S + col];
        if (gk < K && (gn + 4) <= N) {
          float4 f = *(const float4*)(Bw + (long)gk * ldb + gn);
          dst[0] = (_Float16)f.x; dst[1] = (_Float16)f.y;
          dst[2] = (_Float16)f.z; dst[3] = (_Float16)f.w;
        } else {
#pragma unroll
          for (int e = 0; e < 4; ++e) {
            float fv = (gk < K && (gn + e) < N) ? Bw[(long)gk * ldb + gn + e] : 0.f;
            dst[e] = (_Float16)fv;
          }
        }
      }
      __syncthreads();

      v16h af[4], bf[2];
#pragma unroll
      for (int mi = 0; mi < 4; ++mi) af[mi] = load_a_frag(As[0], wm * 64 + mi * 16, lane);
#pragma unroll
      for (int ni = 0; ni < 2; ++ni) bf[ni] = load_b_frag(Bs[0], wn * 32 + ni * 16, lane);
#pragma unroll
      for (int mi = 0; mi < 4; ++mi)
#pragma unroll
        for (int ni = 0; ni < 2; ++ni)
          acc[mi][ni] = __builtin_amdgcn_wmma_f32_16x16x32_f16(
              false, af[mi], false, bf[ni], (short)0, acc[mi][ni], false, false);
      __syncthreads();
    }
  }

  // epilogue: C layout — VGPR r holds row (r + 8*hs), lane&15 is the column
  const int hs = lane >> 4;
#pragma unroll
  for (int mi = 0; mi < 4; ++mi)
#pragma unroll
    for (int ni = 0; ni < 2; ++ni)
#pragma unroll
      for (int r = 0; r < 8; ++r) {
        int gm = m0 + wm * 64 + mi * 16 + r + 8 * hs;
        int gn = n0 + wn * 32 + ni * 16 + (lane & 15);
        if (ALIGNED || (gm < M && gn < N)) {
          float v = alpha * acc[mi][ni][r];
          long idx = (long)gm * ldc + gn;
          if (accFlag) C[idx] += v; else C[idx] = v;
        }
      }
}

// ---------------------------------------------------------------------------
// RoPE in place on [B,S,H,HD]; one thread per (d, d+64) pair
// ---------------------------------------------------------------------------
__global__ void k_rope(float* __restrict__ x) {
  long i = (long)blockIdx.x * blockDim.x + threadIdx.x;   // B*S*H*64 threads
  int d = (int)(i & 63);
  long rest = i >> 6;
  int hh = (int)(rest % HH);
  long bs = rest / HH;
  int s = (int)(bs % SD);
  float inv = __expf(-(float)d * (9.210340372f / 64.f));  // 10000^(-d/64)
  float ang = (float)s * inv;
  float c = __cosf(ang), sn = __sinf(ang);
  long base = (bs * HH + hh) * HDD;
  float x1 = x[base + d], x2 = x[base + d + 64];
  x[base + d]      = x1 * c - x2 * sn;
  x[base + d + 64] = x2 * c + x1 * sn;
}

// ---------------------------------------------------------------------------
// Attention: one 128-thread block per (q-row, head, batch). Two-pass softmax
// with LDS score buffer; causal + attention_mask.
// ---------------------------------------------------------------------------
__global__ __launch_bounds__(128)
void k_attn(const float* __restrict__ q, const float* __restrict__ k,
            const float* __restrict__ v, const int* __restrict__ amask,
            float* __restrict__ ctx) {
  int qi = blockIdx.x, hh = blockIdx.y, b = blockIdx.z;
  int tid = threadIdx.x;
  __shared__ float sc[SD];
  __shared__ float qs[HDD];
  __shared__ float red[128];
  long qbase = (((long)b * SD + qi) * HH + hh) * HDD;
  qs[tid] = q[qbase + tid] * 0.08838834764831845f;   // 1/sqrt(128)
  __syncthreads();

  for (int j = tid; j <= qi; j += 128) {
    const float* kr = k + (((long)b * SD + j) * HH + hh) * HDD;
    float dot = 0.f;
#pragma unroll 8
    for (int d = 0; d < HDD; ++d) dot += qs[d] * kr[d];
    sc[j] = (amask[b * SD + j] > 0) ? dot : -1e9f;
  }
  __syncthreads();

  float mx = -1e30f;
  for (int j = tid; j <= qi; j += 128) mx = fmaxf(mx, sc[j]);
  red[tid] = mx; __syncthreads();
  for (int o = 64; o > 0; o >>= 1) {
    if (tid < o) red[tid] = fmaxf(red[tid], red[tid + o]);
    __syncthreads();
  }
  mx = red[0]; __syncthreads();

  float ls = 0.f;
  for (int j = tid; j <= qi; j += 128) {
    float e = __expf(sc[j] - mx); sc[j] = e; ls += e;
  }
  red[tid] = ls; __syncthreads();
  for (int o = 64; o > 0; o >>= 1) {
    if (tid < o) red[tid] += red[tid + o];
    __syncthreads();
  }
  ls = red[0]; __syncthreads();

  float accv = 0.f;
  const float* vp = v + (((long)b * SD) * HH + hh) * HDD + tid;  // d = tid
  for (int j = 0; j <= qi; ++j) accv += sc[j] * vp[(long)j * HH * HDD];
  ctx[qbase + tid] = accv / ls;
}

// ---------------------------------------------------------------------------
// Elementwise helpers
// ---------------------------------------------------------------------------
__global__ void k_cast16(const float* __restrict__ x, _Float16* __restrict__ o, long n) {
  long i = (long)blockIdx.x * blockDim.x + threadIdx.x;
  if (i < n) o[i] = (_Float16)x[i];
}

__global__ void k_silu_mul(const float* __restrict__ g, const float* __restrict__ u,
                           _Float16* __restrict__ o, long n) {
  long i = (long)blockIdx.x * blockDim.x + threadIdx.x;
  if (i < n) {
    float gv = g[i];
    float s = gv / (1.f + __expf(-gv));
    o[i] = (_Float16)(s * u[i]);
  }
}

// ---------------------------------------------------------------------------
// Regression head: out[b,o] = hn16[last token, :] . Wreg[:, o] + breg[o]
// ---------------------------------------------------------------------------
__global__ __launch_bounds__(256)
void k_head(const _Float16* __restrict__ hn, const float* __restrict__ Wreg,
            const float* __restrict__ breg, float* __restrict__ out) {
  int b = blockIdx.x / OUTD, o = blockIdx.x % OUTD;
  const _Float16* row = hn + ((long)b * SD + SD - 1) * DD;
  __shared__ float red[256];
  float s = 0.f;
  for (int d = threadIdx.x; d < DD; d += 256)
    s += (float)row[d] * Wreg[(long)d * OUTD + o];
  red[threadIdx.x] = s; __syncthreads();
  for (int off = 128; off > 0; off >>= 1) {
    if (threadIdx.x < off) red[threadIdx.x] += red[threadIdx.x + off];
    __syncthreads();
  }
  if (threadIdx.x == 0) out[b * OUTD + o] = red[0] + breg[o];
}

// ---------------------------------------------------------------------------
// Orchestration
// ---------------------------------------------------------------------------
extern "C" void kernel_launch(void* const* d_in, const int* in_sizes, int n_in,
                              void* d_out, int out_size, void* d_ws, size_t ws_size,
                              hipStream_t stream) {
  (void)in_sizes; (void)n_in; (void)out_size; (void)ws_size;

  const int*   ids    = (const int*)d_in[0];
  const int*   amask  = (const int*)d_in[1];
  const float* embed  = (const float*)d_in[2];
  const float* Wq     = (const float*)d_in[3];
  const float* Wk     = (const float*)d_in[4];
  const float* Wv     = (const float*)d_in[5];
  const float* Wo     = (const float*)d_in[6];
  const float* Aq     = (const float*)d_in[7];
  const float* Bq     = (const float*)d_in[8];
  const float* Av     = (const float*)d_in[9];
  const float* Bv     = (const float*)d_in[10];
  const float* norm1  = (const float*)d_in[11];
  const float* norm2  = (const float*)d_in[12];
  const float* Wgate  = (const float*)d_in[13];
  const float* Wup    = (const float*)d_in[14];
  const float* Wdown  = (const float*)d_in[15];
  const float* norm_f = (const float*)d_in[16];
  const float* Wreg   = (const float*)d_in[17];
  const float* breg   = (const float*)d_in[18];
  float* out = (float*)d_out;

  char* ws = (char*)d_ws;
  size_t off = 0;
  auto alloc = [&](size_t bytes) -> char* {
    char* p = ws + off;
    off += (bytes + 255) & ~(size_t)255;
    return p;
  };

  float*    hbuf   = (float*)   alloc((size_t)MTOK * DD * 4);
  _Float16* hn16   = (_Float16*)alloc((size_t)MTOK * DD * 2);
  float*    qb     = (float*)   alloc((size_t)MTOK * DD * 4);
  float*    kb     = (float*)   alloc((size_t)MTOK * DD * 4);
  float*    vb     = (float*)   alloc((size_t)MTOK * DD * 4);
  float*    ctx    = (float*)   alloc((size_t)MTOK * DD * 4);
  float*    lora   = (float*)   alloc((size_t)MTOK * RR * 4);
  _Float16* lora16 = (_Float16*)alloc((size_t)MTOK * RR * 2);
  float*    Gb     = (float*)   alloc((size_t)MTOK * FF * 4);
  float*    Ub     = (float*)   alloc((size_t)MTOK * FF * 4);
  _Float16* act16  = (_Float16*)alloc((size_t)MTOK * FF * 2);

  auto gemmA = [&](const _Float16* A, int lda, const float* Bw, int ldb,
                   float* C, int ldc, int M, int N, int K, float alpha, int accF) {
    dim3 g(N / TN, M / TM);
    k_gemm_f16<true><<<g, 256, 0, stream>>>(A, lda, Bw, ldb, C, ldc, M, N, K, alpha, accF);
  };
  auto gemmU = [&](const _Float16* A, int lda, const float* Bw, int ldb,
                   float* C, int ldc, int M, int N, int K, float alpha, int accF) {
    dim3 g((N + TN - 1) / TN, (M + TM - 1) / TM);
    k_gemm_f16<false><<<g, 256, 0, stream>>>(A, lda, Bw, ldb, C, ldc, M, N, K, alpha, accF);
  };

  // embedding
  {
    long n = (long)MTOK * DD;
    k_embed<<<(unsigned)(n / 256), 256, 0, stream>>>(ids, embed, hbuf);
  }

  for (int l = 0; l < 2; ++l) {
    const float* wq = Wq + (size_t)l * DD * DD;
    const float* wk = Wk + (size_t)l * DD * DD;
    const float* wv = Wv + (size_t)l * DD * DD;
    const float* wo = Wo + (size_t)l * DD * DD;
    const float* aq = Aq + (size_t)l * DD * RR;
    const float* bq = Bq + (size_t)l * RR * DD;
    const float* av = Av + (size_t)l * DD * RR;
    const float* bv = Bv + (size_t)l * RR * DD;
    const float* g1 = norm1 + (size_t)l * DD;
    const float* g2 = norm2 + (size_t)l * DD;
    const float* wg = Wgate + (size_t)l * DD * FF;
    const float* wu = Wup   + (size_t)l * DD * FF;
    const float* wd = Wdown + (size_t)l * FF * DD;

    k_rmsnorm<<<MTOK, 256, 0, stream>>>(hbuf, g1, hn16);

    // q = hn @ Wq + 2.0 * (hn @ Aq) @ Bq
    gemmA(hn16, DD, wq, DD, qb, DD, MTOK, DD, DD, 1.f, 0);
    gemmU(hn16, DD, aq, RR, lora, RR, MTOK, RR, DD, 1.f, 0);
    k_cast16<<<(MTOK * RR + 255) / 256, 256, 0, stream>>>(lora, lora16, (long)MTOK * RR);
    gemmU(lora16, RR, bq, DD, qb, DD, MTOK, DD, RR, 2.0f, 1);

    // k = hn @ Wk
    gemmA(hn16, DD, wk, DD, kb, DD, MTOK, DD, DD, 1.f, 0);

    // v = hn @ Wv + 2.0 * (hn @ Av) @ Bv
    gemmA(hn16, DD, wv, DD, vb, DD, MTOK, DD, DD, 1.f, 0);
    gemmU(hn16, DD, av, RR, lora, RR, MTOK, RR, DD, 1.f, 0);
    k_cast16<<<(MTOK * RR + 255) / 256, 256, 0, stream>>>(lora, lora16, (long)MTOK * RR);
    gemmU(lora16, RR, bv, DD, vb, DD, MTOK, DD, RR, 2.0f, 1);

    // rope on q and k
    {
      long n = (long)BD * SD * HH * 64;
      k_rope<<<(unsigned)(n / 256), 256, 0, stream>>>(qb);
      k_rope<<<(unsigned)(n / 256), 256, 0, stream>>>(kb);
    }

    // attention
    k_attn<<<dim3(SD, HH, BD), 128, 0, stream>>>(qb, kb, vb, amask, ctx);

    // h += ctx @ Wo
    k_cast16<<<(unsigned)(((long)MTOK * DD + 255) / 256), 256, 0, stream>>>(
        ctx, act16, (long)MTOK * DD);
    gemmA(act16, DD, wo, DD, hbuf, DD, MTOK, DD, DD, 1.f, 1);

    // MLP
    k_rmsnorm<<<MTOK, 256, 0, stream>>>(hbuf, g2, hn16);
    gemmA(hn16, DD, wg, FF, Gb, FF, MTOK, FF, DD, 1.f, 0);
    gemmA(hn16, DD, wu, FF, Ub, FF, MTOK, FF, DD, 1.f, 0);
    k_silu_mul<<<(unsigned)(((long)MTOK * FF + 255) / 256), 256, 0, stream>>>(
        Gb, Ub, act16, (long)MTOK * FF);
    gemmA(act16, FF, wd, DD, hbuf, DD, MTOK, DD, FF, 1.f, 1);
  }

  // final norm + head
  k_rmsnorm<<<MTOK, 256, 0, stream>>>(hbuf, norm_f, hn16);
  k_head<<<BD * OUTD, 256, 0, stream>>>(hn16, Wreg, breg, out);
}
